// GATv2Encoder_65515431133657
// MI455X (gfx1250) — matrix operand order repaired and checked by reading, not compiled, not run on hardware
//
#include <hip/hip_runtime.h>

typedef __attribute__((ext_vector_type(8)))  _Float16 v8h;
typedef __attribute__((ext_vector_type(16))) _Float16 v16h;
typedef __attribute__((ext_vector_type(8)))  float    v8f;

static __device__ __forceinline__ v16h cat16(v8h lo, v8h hi) {
  return __builtin_shufflevector(lo, hi, 0,1,2,3,4,5,6,7,8,9,10,11,12,13,14,15);
}

// ordered-key encoding so unsigned atomicMax == float max
static __device__ __forceinline__ unsigned f2key(float f) {
  unsigned u = __float_as_uint(f);
  return (u & 0x80000000u) ? ~u : (u | 0x80000000u);
}
static __device__ __forceinline__ float key2f(unsigned k) {
  unsigned u = (k & 0x80000000u) ? (k ^ 0x80000000u) : ~k;
  return __uint_as_float(u);
}

// ---------------------------------------------------------------------------
// Generic Y[M, NC*16] = X[M,K] @ W[K, NC*16] (+bias) with f16 WMMA, f32 accum.
// Block = 128 threads (4 waves); each wave does a 16-row x (NC*16)-col tile.
// ---------------------------------------------------------------------------
template<int NC>
__global__ void __launch_bounds__(128)
gemm_wmma_kernel(const float* __restrict__ X, const float* __restrict__ W,
                 const float* __restrict__ bias, float* __restrict__ Y,
                 int M, int K, int relu)
{
  __shared__ _Float16 Ws[128 * 136] __attribute__((aligned(16))); // [col][k] transposed
  __shared__ _Float16 Xs[64 * 136]  __attribute__((aligned(16))); // [row][k]
  const int tid = threadIdx.x;
  const int Ncols = NC * 16;
  const int Kpad = K + 8;

  for (int i = tid; i < K * Ncols; i += 128) {
    int k = i / Ncols, n = i - k * Ncols;
    Ws[n * Kpad + k] = (_Float16)W[i];
  }
  const long long rowBase = (long long)blockIdx.x * 64;
  for (int i = tid; i < 64 * K; i += 128) {
    int r = i / K, k = i - r * K;
    long long row = rowBase + r;
    float v = (row < (long long)M) ? X[row * (long long)K + k] : 0.f;
    Xs[r * Kpad + k] = (_Float16)v;
  }
  __syncthreads();

  const int wave = tid >> 5, lane = tid & 31;
  const int ln = lane & 15, half = lane >> 4;
  const int wr = wave * 16;

  v8f acc[NC];
#pragma unroll
  for (int nt = 0; nt < NC; ++nt) acc[nt] = (v8f){0.f,0.f,0.f,0.f,0.f,0.f,0.f,0.f};

  const int kTiles = K >> 5;
  for (int kt = 0; kt < kTiles; ++kt) {
    const int kb = kt * 32;
    // A fragment: row = wr+ln; el[0..7] -> k = kb + half*8 + j ; el[8..15] -> +16
    const _Float16* xp = Xs + (size_t)(wr + ln) * Kpad + kb;
    v8h alo = *(const v8h*)(xp + half * 8);
    v8h ahi = *(const v8h*)(xp + 16 + half * 8);
    v16h a = cat16(alo, ahi);
#pragma unroll
    for (int nt = 0; nt < NC; ++nt) {
      // B fragment: col = nt*16+ln; el[0..7] -> k = kb + half*8 + j ; el[8..15] -> +16
      const _Float16* wp = Ws + (size_t)(nt * 16 + ln) * Kpad + kb + half * 8;
      v8h blo = *(const v8h*)wp;
      v8h bhi = *(const v8h*)(wp + 16);
      v16h b = cat16(blo, bhi);
      acc[nt] = __builtin_amdgcn_wmma_f32_16x16x32_f16(
          false, a, false, b, (short)0, acc[nt], false, false);
    }
  }
#pragma unroll
  for (int nt = 0; nt < NC; ++nt) {
    const int col = nt * 16 + ln;
    float bv = bias ? bias[col] : 0.f;
#pragma unroll
    for (int r = 0; r < 8; ++r) {
      long long row = rowBase + wr + half * 8 + r;
      if (row < (long long)M) {
        float v = acc[nt][r] + bv;
        if (relu) v = v > 0.f ? v : 0.f;
        Y[row * (long long)Ncols + col] = v;
      }
    }
  }
}

// ---------------------------------------------------------------------------
// Fused edge kernel: per wave handles 16 edges.
//   em(16x128) = edge_attr(16x32) @ We(32x128) via 8 WMMAs,
//   m = leaky_relu(em + XL[src] + XR[dst]), logits[e,h] = sum_c m*att[h,c],
//   then atomicMax of ordered-key logits into mxk[dst,h].
// ---------------------------------------------------------------------------
__global__ void __launch_bounds__(256)
edge_logits_kernel(const float* __restrict__ eattr, const int* __restrict__ srcp,
                   const int* __restrict__ dstp, const float* __restrict__ We,
                   const float* __restrict__ att, const float* __restrict__ XL,
                   const float* __restrict__ XR, float* __restrict__ logits,
                   unsigned* __restrict__ mxk, int E)
{
  __shared__ _Float16 WeT[128 * 40] __attribute__((aligned(16))); // [col][k]
  __shared__ float satt[128];
  const int tid = threadIdx.x;
  for (int i = tid; i < 32 * 128; i += 256) {
    int k = i >> 7, col = i & 127;
    WeT[col * 40 + k] = (_Float16)We[i];
  }
  if (tid < 128) satt[tid] = att[tid];
  __syncthreads();

  const int wave = tid >> 5, lane = tid & 31;
  const int ln = lane & 15, half = lane >> 4;
  const long long eb = (long long)blockIdx.x * 128 + wave * 16;
  if (eb >= (long long)E) return;              // wave-uniform exit
  long long erow = eb + ln;
  if (erow >= (long long)E) erow = E - 1;      // clamp; stores are guarded
  const int sid = srcp[erow], did = dstp[erow];

  // A fragment from edge_attr row (f32 -> f16)
  v16h a;
  {
    const float* rp = eattr + (size_t)erow * 32 + half * 8;
    float4 f0 = *(const float4*)(rp);
    float4 f1 = *(const float4*)(rp + 4);
    float4 f2 = *(const float4*)(rp + 16);
    float4 f3 = *(const float4*)(rp + 20);
    a[0]=(_Float16)f0.x;  a[1]=(_Float16)f0.y;  a[2]=(_Float16)f0.z;  a[3]=(_Float16)f0.w;
    a[4]=(_Float16)f1.x;  a[5]=(_Float16)f1.y;  a[6]=(_Float16)f1.z;  a[7]=(_Float16)f1.w;
    a[8]=(_Float16)f2.x;  a[9]=(_Float16)f2.y;  a[10]=(_Float16)f2.z; a[11]=(_Float16)f2.w;
    a[12]=(_Float16)f3.x; a[13]=(_Float16)f3.y; a[14]=(_Float16)f3.z; a[15]=(_Float16)f3.w;
  }

  float plog[8][4];
#pragma unroll
  for (int r = 0; r < 8; ++r)
#pragma unroll
    for (int h = 0; h < 4; ++h) plog[r][h] = 0.f;

#pragma unroll
  for (int nt = 0; nt < 8; ++nt) {
    const int col = nt * 16 + ln;
    const _Float16* wp = WeT + (size_t)col * 40 + half * 8;
    v8h blo = *(const v8h*)wp;
    v8h bhi = *(const v8h*)(wp + 16);
    v16h b = cat16(blo, bhi);
    v8f c = {};
    c = __builtin_amdgcn_wmma_f32_16x16x32_f16(false, a, false, b, (short)0, c, false, false);
    const int head = nt >> 1;
    const float attw = satt[head * 32 + (nt & 1) * 16 + ln];
#pragma unroll
    for (int r = 0; r < 8; ++r) {
      const int q = half * 8 + r;                 // edge row in this 16-edge tile
      int s = __shfl(sid, q, 32);                 // lane q holds row q's ids
      int d = __shfl(did, q, 32);
      float m = c[r] + XL[(size_t)s * 128 + col] + XR[(size_t)d * 128 + col];
      m = (m > 0.f) ? m : 0.2f * m;               // leaky_relu
      plog[r][head] += m * attw;
    }
  }
  // reduce the 16 columns held per half-group (lanes 0-15 rows 0-7, 16-31 rows 8-15)
#pragma unroll
  for (int mask = 1; mask < 16; mask <<= 1) {
#pragma unroll
    for (int r = 0; r < 8; ++r)
#pragma unroll
      for (int h = 0; h < 4; ++h)
        plog[r][h] += __shfl_xor(plog[r][h], mask, 32);
  }
#pragma unroll
  for (int r = 0; r < 8; ++r) {
    if (ln == r) {
      long long e2 = eb + half * 8 + r;
      if (e2 < (long long)E) {
        int d2 = dstp[e2];
#pragma unroll
        for (int h = 0; h < 4; ++h) {
          float lv = plog[r][h];
          logits[(size_t)e2 * 4 + h] = lv;
          atomicMax(mxk + (size_t)d2 * 4 + h, f2key(lv));
        }
      }
    }
  }
}

// ex = exp(logit - max[dst]); den[dst] += ex  (overwrites logits with ex)
__global__ void softmax_kernel(float* __restrict__ lg, const int* __restrict__ dstp,
                               const unsigned* __restrict__ mxk,
                               float* __restrict__ den, int E)
{
  int i = blockIdx.x * 256 + threadIdx.x;
  if (i >= E * 4) return;
  int e = i >> 2, h = i & 3;
  int d = dstp[e];
  float mx = key2f(mxk[(size_t)d * 4 + h]);
  float ex = __expf(lg[i] - mx);
  lg[i] = ex;
  atomicAdd(den + (size_t)d * 4 + h, ex);
}

// agg[dst] += XL[src] * (ex/den[dst]); one wave per edge, 128 cols in 4 passes
__global__ void __launch_bounds__(256)
aggregate_kernel(const float* __restrict__ XL, const float* __restrict__ ex,
                 const float* __restrict__ den, const int* __restrict__ srcp,
                 const int* __restrict__ dstp, float* __restrict__ agg, int E)
{
  const int wave = threadIdx.x >> 5, lane = threadIdx.x & 31;
  long long e = (long long)blockIdx.x * 8 + wave;
  if (e >= (long long)E) return;
  const int s = srcp[e], d = dstp[e];
  const float* xlr = XL + (size_t)s * 128;
  float* ag = agg + (size_t)d * 128;
#pragma unroll
  for (int it = 0; it < 4; ++it) {
    float alpha = ex[(size_t)e * 4 + it] / den[(size_t)d * 4 + it];
    int c = it * 32 + lane;
    atomicAdd(ag + c, xlr[c] * alpha);
  }
}

__global__ void init_kernel(float* __restrict__ agg, float* __restrict__ den,
                            unsigned* __restrict__ mxk, size_t nf, int N)
{
  size_t i = (size_t)blockIdx.x * 256 + threadIdx.x;
  if (i < nf) agg[i] = 0.f;
  if (i < (size_t)N * 4) { den[i] = 0.f; mxk[i] = 0u; }
}

__global__ void bias_relu_kernel(float* __restrict__ h, const float* __restrict__ b, size_t n)
{
  size_t i = (size_t)blockIdx.x * 256 + threadIdx.x;
  if (i < n) {
    float v = h[i] + b[i & 127];
    h[i] = v > 0.f ? v : 0.f;
  }
}

extern "C" void kernel_launch(void* const* d_in, const int* in_sizes, int n_in,
                              void* d_out, int out_size, void* d_ws, size_t ws_size,
                              hipStream_t stream) {
  (void)n_in; (void)out_size; (void)ws_size;
  const float* x     = (const float*)d_in[0];
  const int*   srcp  = (const int*)d_in[1];
  const int*   dstp  = (const int*)d_in[2];
  const float* eattr = (const float*)d_in[3];
  const float* Wl1 = (const float*)d_in[4];
  const float* bl1 = (const float*)d_in[5];
  const float* Wr1 = (const float*)d_in[6];
  const float* We1 = (const float*)d_in[7];
  const float* att1= (const float*)d_in[8];
  const float* b1  = (const float*)d_in[9];
  const float* Wl2 = (const float*)d_in[10];
  const float* bl2 = (const float*)d_in[11];
  const float* Wr2 = (const float*)d_in[12];
  const float* We2 = (const float*)d_in[13];
  const float* att2= (const float*)d_in[14];
  const float* b2  = (const float*)d_in[15];
  const float* Wo1 = (const float*)d_in[16];
  const float* bo1 = (const float*)d_in[17];
  const float* Wo2 = (const float*)d_in[18];
  const float* bo2 = (const float*)d_in[19];

  const int N = in_sizes[0] / 128;
  const int E = in_sizes[1];
  const size_t NF = (size_t)N * 128;

  float* ws  = (float*)d_ws;
  float* XL  = ws;                 // [N,128] (also reused as T [N,32])
  float* XR  = ws + NF;            // [N,128]
  float* AGG = ws + 2 * NF;        // [N,128] aggregation / hidden
  float* LOG = ws + 3 * NF;        // [E,4] logits -> exp
  unsigned* MXK = (unsigned*)(LOG + (size_t)E * 4); // [N,4] ordered-key max
  float* DEN = (float*)(MXK + (size_t)N * 4);       // [N,4]

  const unsigned gGemm = (unsigned)((N + 63) / 64);
  const unsigned gElem = (unsigned)((NF + 255) / 256);
  const unsigned gEdge = (unsigned)((E + 127) / 128);
  const unsigned gSm   = (unsigned)(((size_t)E * 4 + 255) / 256);
  const unsigned gAgg  = (unsigned)((E + 7) / 8);

  // ---- layer 1 ----
  gemm_wmma_kernel<8><<<gGemm, 128, 0, stream>>>(x, Wl1, bl1, XL, N, 128, 0);
  gemm_wmma_kernel<8><<<gGemm, 128, 0, stream>>>(x, Wr1, nullptr, XR, N, 128, 0);
  init_kernel<<<gElem, 256, 0, stream>>>(AGG, DEN, MXK, NF, N);
  edge_logits_kernel<<<gEdge, 256, 0, stream>>>(eattr, srcp, dstp, We1, att1, XL, XR, LOG, MXK, E);
  softmax_kernel<<<gSm, 256, 0, stream>>>(LOG, dstp, MXK, DEN, E);
  aggregate_kernel<<<gAgg, 256, 0, stream>>>(XL, LOG, DEN, srcp, dstp, AGG, E);
  bias_relu_kernel<<<gElem, 256, 0, stream>>>(AGG, b1, NF);   // AGG = h1

  // ---- layer 2 (consume h1 before re-zeroing AGG) ----
  gemm_wmma_kernel<8><<<gGemm, 128, 0, stream>>>(AGG, Wl2, bl2, XL, N, 128, 0);
  gemm_wmma_kernel<8><<<gGemm, 128, 0, stream>>>(AGG, Wr2, nullptr, XR, N, 128, 0);
  init_kernel<<<gElem, 256, 0, stream>>>(AGG, DEN, MXK, NF, N);
  edge_logits_kernel<<<gEdge, 256, 0, stream>>>(eattr, srcp, dstp, We2, att2, XL, XR, LOG, MXK, E);
  softmax_kernel<<<gSm, 256, 0, stream>>>(LOG, dstp, MXK, DEN, E);
  aggregate_kernel<<<gAgg, 256, 0, stream>>>(XL, LOG, DEN, srcp, dstp, AGG, E);
  bias_relu_kernel<<<gElem, 256, 0, stream>>>(AGG, b2, NF);   // AGG = h2

  // ---- output MLP: (h2 @ Wo1 + bo1) @ Wo2 + bo2 ----
  gemm_wmma_kernel<2><<<gGemm, 128, 0, stream>>>(AGG, Wo1, bo1, XL, N, 128, 0);        // T [N,32]
  gemm_wmma_kernel<4><<<gGemm, 128, 0, stream>>>(XL, Wo2, bo2, (float*)d_out, N, 32, 0); // [N,64]
}
